// Basicblock_51092930953591
// MI455X (gfx1250) — 8-bit matrix operand order repaired and checked by reading, compiled, not run on hardware
//
#include <hip/hip_runtime.h>
#include <hip/hip_bf16.h>

typedef int  v8i __attribute__((ext_vector_type(8)));
typedef int  v4i __attribute__((ext_vector_type(4)));

#define CCH   256
#define HW    3136
#define NB    32
#define NPIX  (NB*HW)
#define NELEM ((size_t)NB*CCH*HW)

__global__ void pack_w3x3(const float* __restrict__ w, signed char* __restrict__ frag) {
    int t = blockIdx.x * 256 + threadIdx.x;
    int i    = t & 31;
    int lane = (t >> 5) & 31;
    int rest = t >> 10;
    int ot   = rest & 15;
    int kb   = rest >> 4;
    int v = i >> 2, b = i & 3;
    int koff = ((v & 4) ? 32 : 0) + ((v & 2) ? 8 : 0) + ((v & 1) ? 4 : 0) + b
             + ((lane & 16) ? 16 : 0);
    int k   = kb * 64 + koff;
    int o   = ot * 16 + (lane & 15);
    int tap = k >> 8;
    int c   = k & 255;
    float wv = w[((size_t)(o * CCH + c)) * 9 + tap];
    frag[t] = (wv > 0.f) ? 1 : ((wv < 0.f) ? -1 : 0);
}

__global__ void pack_wres(const float* __restrict__ w, signed char* __restrict__ frag) {
    int t = blockIdx.x * 256 + threadIdx.x;
    int i    = t & 31;
    int lane = (t >> 5) & 31;
    int rest = t >> 10;
    int ot   = rest & 15;
    int kb   = rest >> 4;
    int v = i >> 2, b = i & 3;
    int koff = ((v & 4) ? 32 : 0) + ((v & 2) ? 8 : 0) + ((v & 1) ? 4 : 0) + b
             + ((lane & 16) ? 16 : 0);
    int c = kb * 64 + koff;
    int o = ot * 16 + (lane & 15);
    float wv = w[(size_t)o * CCH + c];
    frag[t] = (wv > 0.f) ? 1 : ((wv < 0.f) ? -1 : 0);
}

__global__ void alphas(const float* __restrict__ w3, const float* __restrict__ wr,
                       float* __restrict__ a1, float* __restrict__ a2) {
    int o = threadIdx.x;
    if (blockIdx.x == 0) {
        const float* p = w3 + (size_t)o * 2304;
        float s = 0.f;
        for (int i = 0; i < 2304; ++i) s += fabsf(p[i]);
        a1[o] = s * (1.0f / 2304.0f);
    } else {
        const float* p = wr + (size_t)o * 256;
        float s = 0.f;
        for (int i = 0; i < 256; ++i) s += fabsf(p[i]);
        a2[o] = s * (1.0f / 256.0f);
    }
}

__global__ void binact(const float* __restrict__ x, const float* __restrict__ bias,
                       signed char* __restrict__ act) {
    int bid = blockIdx.x;
    int pt = bid % 98;  bid /= 98;
    int ct = bid % 8;   int n = bid / 8;
    int tx = threadIdx.x & 31;
    int ty = threadIdx.x >> 5;
    __shared__ signed char tileS[32][33];
    int c0 = ct * 32, p0 = pt * 32;
    #pragma unroll
    for (int cc = ty; cc < 32; cc += 8) {
        int c = c0 + cc;
        float v = x[((size_t)(n * CCH + c)) * HW + p0 + tx] + bias[c];
        tileS[cc][tx] = (v > 0.f) ? 1 : ((v < 0.f) ? -1 : 0);
    }
    __syncthreads();
    #pragma unroll
    for (int pp = ty; pp < 32; pp += 8) {
        act[((size_t)(n * HW + p0 + pp)) * CCH + c0 + tx] = tileS[tx][pp];
    }
}

template <int NT>
__global__ void __launch_bounds__(512, 1)
bconv(const signed char* __restrict__ act,
      const signed char* __restrict__ wfrag,
      const float* __restrict__ alpha,
      float* __restrict__ y) {
    const int tid   = threadIdx.x;
    const int lane  = tid & 31;
    const int wv    = tid >> 5;
    const int ptile = blockIdx.x * 8 + (wv >> 1);
    const int ohalf = wv & 1;
    const int p0    = ptile << 4;
    const int n     = p0 / HW;
    const int hw0   = p0 % HW;
    const int lp    = lane & 15;
    const int hw    = hw0 + lp;
    const int h     = hw / 56, w_ = hw % 56;
    const int koffB = (lane & 16) ? 16 : 0;
    const int NKB   = NT * 4;

    __shared__ signed char ldsA[2][16384];

    v8i acc[8] = {};

    {
        const v4i* src = (const v4i*)(wfrag + tid * 32);
        v4i* dst = (v4i*)(ldsA[0] + tid * 32);
        v4i r0 = src[0], r1 = src[1];
        dst[0] = r0; dst[1] = r1;
    }
    __syncthreads();

    #pragma unroll 1
    for (int kb = 0; kb < NKB; ++kb) {
        v4i r0 = {}, r1 = {};
        if (kb + 1 < NKB) {
            const v4i* src = (const v4i*)(wfrag + (size_t)(kb + 1) * 16384 + tid * 32);
            r0 = src[0]; r1 = src[1];
        }

        const int tap = kb >> 2, cb = kb & 3;
        int hh = h, ww = w_;
        bool valid = true;
        if (NT == 9) {
            hh = h + tap / 3 - 1;
            ww = w_ + tap % 3 - 1;
            valid = ((unsigned)hh < 56u) && ((unsigned)ww < 56u);
        }
        const signed char* bptr =
            act + (((size_t)(n * HW + hh * 56 + ww)) << 8) + cb * 64 + koffB;
        __builtin_prefetch(bptr + 56 * 256, 0, 0);
        v4i b0 = {}, b1 = {};
        if (valid) {
            b0 = *(const v4i*)bptr;
            b1 = *(const v4i*)(bptr + 32);
        }
        v8i B;
        B[0] = b0[0]; B[1] = b0[1]; B[2] = b0[2]; B[3] = b0[3];
        B[4] = b1[0]; B[5] = b1[1]; B[6] = b1[2]; B[7] = b1[3];

        const signed char* aL = ldsA[kb & 1] + ohalf * 8192 + lane * 32;
        v8i A0 = *(const v8i*)(aL + 0 * 1024);
        v8i A1 = *(const v8i*)(aL + 1 * 1024);
        v8i A2 = *(const v8i*)(aL + 2 * 1024);
        v8i A3 = *(const v8i*)(aL + 3 * 1024);
        v8i A4 = *(const v8i*)(aL + 4 * 1024);
        v8i A5 = *(const v8i*)(aL + 5 * 1024);
        v8i A6 = *(const v8i*)(aL + 6 * 1024);
        v8i A7 = *(const v8i*)(aL + 7 * 1024);
        acc[0] = __builtin_amdgcn_wmma_i32_16x16x64_iu8(true, A0, true, B, acc[0], false, false);
        acc[1] = __builtin_amdgcn_wmma_i32_16x16x64_iu8(true, A1, true, B, acc[1], false, false);
        acc[2] = __builtin_amdgcn_wmma_i32_16x16x64_iu8(true, A2, true, B, acc[2], false, false);
        acc[3] = __builtin_amdgcn_wmma_i32_16x16x64_iu8(true, A3, true, B, acc[3], false, false);
        acc[4] = __builtin_amdgcn_wmma_i32_16x16x64_iu8(true, A4, true, B, acc[4], false, false);
        acc[5] = __builtin_amdgcn_wmma_i32_16x16x64_iu8(true, A5, true, B, acc[5], false, false);
        acc[6] = __builtin_amdgcn_wmma_i32_16x16x64_iu8(true, A6, true, B, acc[6], false, false);
        acc[7] = __builtin_amdgcn_wmma_i32_16x16x64_iu8(true, A7, true, B, acc[7], false, false);
#if __has_builtin(__builtin_amdgcn_sched_group_barrier)
        __builtin_amdgcn_sched_group_barrier(0x0100, 16, 0);
        __builtin_amdgcn_sched_group_barrier(0x0008, 8, 0);
#endif

        if (kb + 1 < NKB) {
            __syncthreads();
            v4i* dst = (v4i*)(ldsA[(kb + 1) & 1] + tid * 32);
            dst[0] = r0; dst[1] = r1;
            __syncthreads();
        }
    }

    const int orow = (lane & 16) ? 8 : 0;
    #pragma unroll
    for (int ot = 0; ot < 8; ++ot) {
        #pragma unroll
        for (int j = 0; j < 8; ++j) {
            int o = ohalf * 128 + ot * 16 + j + orow;
            y[((size_t)(n * CCH + o)) * HW + hw0 + lp] = (float)acc[ot][j] * alpha[o];
        }
    }
}

__global__ void chanstats(const float* __restrict__ y,
                          float* __restrict__ mean, float* __restrict__ invstd) {
    int c = blockIdx.x, tid = threadIdx.x;
    float s = 0.f, s2 = 0.f;
    for (int n = 0; n < NB; ++n) {
        const float* p = y + ((size_t)(n * CCH + c)) * HW;
        for (int i = tid; i < HW; i += 256) { float v = p[i]; s += v; s2 += v * v; }
    }
    __shared__ float ls[256], ls2[256];
    ls[tid] = s; ls2[tid] = s2;
    __syncthreads();
    for (int off = 128; off > 0; off >>= 1) {
        if (tid < off) { ls[tid] += ls[tid + off]; ls2[tid] += ls2[tid + off]; }
        __syncthreads();
    }
    if (tid == 0) {
        float m   = ls[0] * (1.0f / (float)NPIX);
        float var = ls2[0] * (1.0f / (float)NPIX) - m * m;
        mean[c]   = m;
        invstd[c] = rsqrtf(var + 1e-5f);
    }
}

__global__ void fuse1(const float* __restrict__ x,  const float* __restrict__ y1,
                      const float* __restrict__ mean, const float* __restrict__ invstd,
                      const float* __restrict__ g,  const float* __restrict__ bb,
                      const float* __restrict__ b1_2, const float* __restrict__ pa,
                      const float* __restrict__ b1_3, const float* __restrict__ b2_1,
                      float* __restrict__ xmid, signed char* __restrict__ act2) {
    int bid = blockIdx.x;
    int pt = bid % 98;  bid /= 98;
    int ct = bid % 8;   int n = bid / 8;
    int tx = threadIdx.x & 31;
    int ty = threadIdx.x >> 5;
    __shared__ signed char tileS[32][33];
    int c0 = ct * 32, p0 = pt * 32;
    #pragma unroll
    for (int cc = ty; cc < 32; cc += 8) {
        int c = c0 + cc;
        size_t idx = ((size_t)(n * CCH + c)) * HW + p0 + tx;
        float t = (y1[idx] - mean[c]) * invstd[c] * g[c] + bb[c];
        float v = x[idx] + t + b1_2[c];
        v = (v >= 0.f) ? v : pa[c] * v;
        v += b1_3[c];
        xmid[idx] = v;
        float sv = v + b2_1[c];
        tileS[cc][tx] = (sv > 0.f) ? 1 : ((sv < 0.f) ? -1 : 0);
    }
    __syncthreads();
    #pragma unroll
    for (int pp = ty; pp < 32; pp += 8) {
        act2[((size_t)(n * HW + p0 + pp)) * CCH + c0 + tx] = tileS[tx][pp];
    }
}

__global__ void fuse2(const float* __restrict__ y2, const float* __restrict__ mean,
                      const float* __restrict__ invstd, const float* __restrict__ g,
                      const float* __restrict__ bb, const float* __restrict__ b2_2,
                      const float* __restrict__ pa, const float* __restrict__ b2_3,
                      float* __restrict__ out) {
    size_t idx = (size_t)blockIdx.x * 256 + threadIdx.x;
    int c = (int)((idx / HW) & (CCH - 1));
    float t = (y2[idx] - mean[c]) * invstd[c] * g[c] + bb[c];
    float v = t + out[idx] + b2_2[c];
    v = (v >= 0.f) ? v : pa[c] * v;
    out[idx] = v + b2_3[c];
}

extern "C" void kernel_launch(void* const* d_in, const int* in_sizes, int n_in,
                              void* d_out, int out_size, void* d_ws, size_t ws_size,
                              hipStream_t stream) {
    const float* x        = (const float*)d_in[0];
    const float* b1_1     = (const float*)d_in[1];
    const float* w3x3     = (const float*)d_in[2];
    const float* bn1_g    = (const float*)d_in[3];
    const float* bn1_b    = (const float*)d_in[4];
    const float* b1_2     = (const float*)d_in[5];
    const float* prelu1_a = (const float*)d_in[6];
    const float* b1_3     = (const float*)d_in[7];
    const float* b2_1     = (const float*)d_in[8];
    const float* wres     = (const float*)d_in[9];
    const float* bn2_g    = (const float*)d_in[10];
    const float* bn2_b    = (const float*)d_in[11];
    const float* b2_2     = (const float*)d_in[12];
    const float* prelu2_a = (const float*)d_in[13];
    const float* b2_3     = (const float*)d_in[14];
    float* out = (float*)d_out;

    char* ws = (char*)d_ws;
    size_t off = 0;
    auto carve = [&](size_t bytes) {
        char* p = ws + off;
        off = (off + bytes + 255) & ~(size_t)255;
        return p;
    };
    signed char* act1   = (signed char*)carve(NELEM);
    signed char* act2   = (signed char*)carve(NELEM);
    float*       ybuf   = (float*)carve(NELEM * 4);
    signed char* w1frag = (signed char*)carve(36 * 16 * 1024);
    signed char* w2frag = (signed char*)carve(4 * 16 * 1024);
    float* alpha1 = (float*)carve(256 * 4);
    float* alpha2 = (float*)carve(256 * 4);
    float* mean1  = (float*)carve(256 * 4);
    float* istd1  = (float*)carve(256 * 4);
    float* mean2  = (float*)carve(256 * 4);
    float* istd2  = (float*)carve(256 * 4);
    (void)ws_size; (void)in_sizes; (void)n_in; (void)out_size;

    pack_w3x3<<<2304, 256, 0, stream>>>(w3x3, w1frag);
    pack_wres<<<256, 256, 0, stream>>>(wres, w2frag);
    alphas<<<2, 256, 0, stream>>>(w3x3, wres, alpha1, alpha2);

    binact<<<32 * 8 * 98, 256, 0, stream>>>(x, b1_1, act1);
    bconv<9><<<784, 512, 0, stream>>>(act1, w1frag, alpha1, ybuf);
    chanstats<<<256, 256, 0, stream>>>(ybuf, mean1, istd1);
    fuse1<<<32 * 8 * 98, 256, 0, stream>>>(x, ybuf, mean1, istd1, bn1_g, bn1_b,
                                           b1_2, prelu1_a, b1_3, b2_1, out, act2);

    bconv<1><<<784, 512, 0, stream>>>(act2, w2frag, alpha2, ybuf);
    chanstats<<<256, 256, 0, stream>>>(ybuf, mean2, istd2);
    fuse2<<<100352, 256, 0, stream>>>(ybuf, mean2, istd2, bn2_g, bn2_b,
                                      b2_2, prelu2_a, b2_3, out);
}
